// NER_LSTMNet_25718264169190
// MI455X (gfx1250) — matrix operand order repaired and checked by reading, not compiled, not run on hardware
//
#include <hip/hip_runtime.h>
#include <stdint.h>

// ---------------------------------------------------------------------------
// Types for CDNA5 WMMA (gfx1250, wave32)
// ---------------------------------------------------------------------------
typedef __attribute__((ext_vector_type(16))) __bf16 v16bf;
typedef __attribute__((ext_vector_type(8)))  float  v8f;

union BF16x16 {
  v16bf v;
  uint4 u[2];
  unsigned short s[16];
};

__device__ __forceinline__ unsigned short f2bf(float f) {
  unsigned int u = __float_as_uint(f);
  u += 0x7FFFu + ((u >> 16) & 1u);   // round-to-nearest-even
  return (unsigned short)(u >> 16);
}

__device__ __forceinline__ float sigmoid_(float x) {
  return 1.0f / (1.0f + __expf(-x));
}

// ---------------------------------------------------------------------------
// Fused LSTM cell step:  gates = [x | h_prev] @ Wcat^T + bias ; c,h update.
//   MODE 0: x-part gathered from fp32 embedding table (encoder layer 0)
//   MODE 1: x-part is a [256 x 512] bf16 activation matrix (enc1 / dec1)
//   MODE 2: x is a one-hot(sel[b]) of size 64 -> column gather of wcol
// Wt is pre-transposed bf16: Wt[k*2048 + n] = W[n][k], rows 0..Kx-1 = Wih^T,
// rows Kx..Kx+511 = Whh^T.  Grid: 64 blocks x 256 threads = 512 waves,
// each wave computes 4 gate tiles (16x16) for one (m,h-col) tile.
// ---------------------------------------------------------------------------
#define WMMA_GATES(KROW)                                                      \
  {                                                                           \
    const unsigned short* brow_ = Wt + (size_t)(KROW)*2048 + n_base + lrow;   \
    _Pragma("unroll")                                                         \
    for (int g_ = 0; g_ < 4; ++g_) {                                          \
      BF16x16 Bf_;                                                            \
      _Pragma("unroll")                                                       \
      for (int s_ = 0; s_ < 16; ++s_) {                                       \
        const int kk_ = (s_ < 8) ? (hlf * 8 + s_) : (16 + hlf * 8 + s_ - 8);  \
        Bf_.s[s_] = brow_[(size_t)kk_ * 2048 + g_ * 512];                     \
      }                                                                       \
      acc[g_] = __builtin_amdgcn_wmma_f32_16x16x32_bf16(                      \
          false, A.v, false, Bf_.v, (short)0, acc[g_], false, false);         \
    }                                                                         \
  }

template <int MODE>
__global__ __launch_bounds__(256) void lstm_cell(
    const unsigned short* __restrict__ Wt,     // [(Kx+512) x 2048] bf16
    const float*          __restrict__ bias,   // [2048] combined bih+bhh
    const unsigned short* __restrict__ xa,     // MODE1: [256 x 512] bf16
    const float*          __restrict__ emb,    // MODE0: [V x 256] fp32
    const int*            __restrict__ tok,    // MODE0: [256 x 128] int
    int t,                                     // MODE0: time index
    const float*          __restrict__ wcol,   // MODE2: dec_Wih0 [2048 x 64]
    const int*            __restrict__ sel,    // MODE2: [256] argmax feedback
    const unsigned short* __restrict__ hprev,  // [256 x 512] bf16
    float*                __restrict__ c,      // [256 x 512] in/out
    unsigned short*       __restrict__ hout_bf,// [256 x 512]
    float*                __restrict__ hout_f) // [256 x 512]
{
  constexpr int Kx = (MODE == 0) ? 256 : ((MODE == 1) ? 512 : 0);
  const int lane   = threadIdx.x & 31;
  const int hlf    = lane >> 4;     // 0 or 1
  const int lrow   = lane & 15;
  const int wid    = (blockIdx.x * 256 + threadIdx.x) >> 5;  // 0..511
  const int m_base = (wid >> 5) << 4;   // 16 m-tiles (B=256)
  const int n_base = (wid & 31) << 4;   // 32 h-col tiles (H=512)

  v8f acc[4] = {};   // i, f, g, o accumulators (16x16 f32 each)
  BF16x16 A;

  // ---- x contribution -----------------------------------------------------
  if constexpr (MODE == 0) {
    const int token = tok[(m_base + lrow) * 128 + t];
    const float* er = emb + (size_t)token * 256;
    for (int k = 0; k < Kx; k += 32) {
      const int k0 = k + hlf * 8;
      const int k1 = k + 16 + hlf * 8;
      float4 f0 = *(const float4*)(er + k0);
      float4 f1 = *(const float4*)(er + k0 + 4);
      float4 f2 = *(const float4*)(er + k1);
      float4 f3 = *(const float4*)(er + k1 + 4);
      A.s[0]  = f2bf(f0.x); A.s[1]  = f2bf(f0.y); A.s[2]  = f2bf(f0.z); A.s[3]  = f2bf(f0.w);
      A.s[4]  = f2bf(f1.x); A.s[5]  = f2bf(f1.y); A.s[6]  = f2bf(f1.z); A.s[7]  = f2bf(f1.w);
      A.s[8]  = f2bf(f2.x); A.s[9]  = f2bf(f2.y); A.s[10] = f2bf(f2.z); A.s[11] = f2bf(f2.w);
      A.s[12] = f2bf(f3.x); A.s[13] = f2bf(f3.y); A.s[14] = f2bf(f3.z); A.s[15] = f2bf(f3.w);
      WMMA_GATES(k);
    }
  }
  if constexpr (MODE == 1) {
    const unsigned short* ar = xa + (size_t)(m_base + lrow) * 512;
    for (int k = 0; k < Kx; k += 32) {
      A.u[0] = *(const uint4*)(ar + k + hlf * 8);
      A.u[1] = *(const uint4*)(ar + k + 16 + hlf * 8);
      WMMA_GATES(k);
    }
  }

  // ---- h_prev contribution ------------------------------------------------
  {
    const unsigned short* hr = hprev + (size_t)(m_base + lrow) * 512;
    for (int k = 0; k < 512; k += 32) {
      A.u[0] = *(const uint4*)(hr + k + hlf * 8);
      A.u[1] = *(const uint4*)(hr + k + 16 + hlf * 8);
      WMMA_GATES(Kx + k);
    }
  }

  // ---- epilogue: bias, (one-hot column), nonlinearities, cell update ------
  const int n = n_base + lrow;
#pragma unroll
  for (int r = 0; r < 8; ++r) {
    const int m = m_base + r + hlf * 8;
    float pi = acc[0][r] + bias[n];
    float pf = acc[1][r] + bias[n + 512];
    float pg = acc[2][r] + bias[n + 1024];
    float po = acc[3][r] + bias[n + 1536];
    if constexpr (MODE == 2) {
      const int s = sel[m];
      pi += wcol[(size_t)n * 64 + s];
      pf += wcol[(size_t)(n + 512) * 64 + s];
      pg += wcol[(size_t)(n + 1024) * 64 + s];
      po += wcol[(size_t)(n + 1536) * 64 + s];
    }
    const float ig = sigmoid_(pi);
    const float fg = sigmoid_(pf);
    const float gg = tanhf(pg);
    const float og = sigmoid_(po);
    const size_t off = (size_t)m * 512 + n;
    const float cv = fg * c[off] + ig * gg;
    const float hv = og * tanhf(cv);
    c[off]       = cv;
    hout_f[off]  = hv;
    hout_bf[off] = f2bf(hv);
  }
}

// ---------------------------------------------------------------------------
// Per-step readout: logits = h1 @ fc_W[t]^T + fc_b[t]; softmax; argmax->sel.
// One block per batch row, 64 threads (one per output class).
// ---------------------------------------------------------------------------
__global__ __launch_bounds__(64) void readout_kernel(
    const float* __restrict__ h1, const float* __restrict__ fcW,
    const float* __restrict__ fcb, float* __restrict__ out,
    int* __restrict__ sel, int t)
{
  const int b = blockIdx.x;
  const int o = threadIdx.x;
  const float4* h4 = (const float4*)(h1 + (size_t)b * 512);
  const float4* w4 = (const float4*)(fcW + ((size_t)t * 64 + o) * 512);
  float acc = fcb[t * 64 + o];
#pragma unroll 8
  for (int k = 0; k < 128; ++k) {
    float4 a = h4[k], w = w4[k];
    acc += a.x * w.x + a.y * w.y + a.z * w.z + a.w * w.w;
  }
  __shared__ float sval[64];
  __shared__ int   sidx[64];
  __shared__ float ssum[64];
  sval[o] = acc; sidx[o] = o;
  __syncthreads();
#pragma unroll
  for (int s2 = 32; s2 > 0; s2 >>= 1) {
    if (o < s2) {
      float v2 = sval[o + s2]; int i2 = sidx[o + s2];
      if (v2 > sval[o] || (v2 == sval[o] && i2 < sidx[o])) { sval[o] = v2; sidx[o] = i2; }
    }
    __syncthreads();
  }
  const float mx = sval[0];
  const int   am = sidx[0];
  const float e  = __expf(acc - mx);
  ssum[o] = e;
  __syncthreads();
#pragma unroll
  for (int s2 = 32; s2 > 0; s2 >>= 1) {
    if (o < s2) ssum[o] += ssum[o + s2];
    __syncthreads();
  }
  out[((size_t)b * 128 + t) * 64 + o] = e / ssum[0];
  if (o == 0) sel[b] = am;
}

// ---------------------------------------------------------------------------
// Setup kernels
// ---------------------------------------------------------------------------
__global__ __launch_bounds__(256) void wt_convert_kernel(
    const float* __restrict__ W, unsigned short* __restrict__ Wt,
    int K, int rowOff, int total)
{
  const int i = blockIdx.x * 256 + threadIdx.x;
  if (i >= total) return;
  const int n = i / K;
  const int k = i - n * K;
  Wt[(size_t)(rowOff + k) * 2048 + n] = f2bf(W[i]);   // transpose to [K x 2048]
}

__global__ __launch_bounds__(256) void bias_combine_kernel(
    const float* __restrict__ a, const float* __restrict__ b,
    float* __restrict__ o2)
{
  const int i = blockIdx.x * 256 + threadIdx.x;
  if (i < 2048) o2[i] = a[i] + b[i];
}

__global__ __launch_bounds__(256) void fill_zero_kernel(
    unsigned int* __restrict__ p, long long n)
{
  const long long i = (long long)blockIdx.x * 256 + threadIdx.x;
  if (i < n) p[i] = 0u;
}

// ---------------------------------------------------------------------------
// Host-side launcher
// ---------------------------------------------------------------------------
extern "C" void kernel_launch(void* const* d_in, const int* in_sizes, int n_in,
                              void* d_out, int out_size, void* d_ws, size_t ws_size,
                              hipStream_t stream) {
  (void)in_sizes; (void)n_in; (void)out_size; (void)ws_size;
  const int*   x     = (const int*)d_in[0];
  const float* emb   = (const float*)d_in[1];
  const float* eWih0 = (const float*)d_in[2];
  const float* eWhh0 = (const float*)d_in[3];
  const float* ebih0 = (const float*)d_in[4];
  const float* ebhh0 = (const float*)d_in[5];
  const float* eWih1 = (const float*)d_in[6];
  const float* eWhh1 = (const float*)d_in[7];
  const float* ebih1 = (const float*)d_in[8];
  const float* ebhh1 = (const float*)d_in[9];
  const float* dWih0 = (const float*)d_in[10];
  const float* dWhh0 = (const float*)d_in[11];
  const float* dbih0 = (const float*)d_in[12];
  const float* dbhh0 = (const float*)d_in[13];
  const float* dWih1 = (const float*)d_in[14];
  const float* dWhh1 = (const float*)d_in[15];
  const float* dbih1 = (const float*)d_in[16];
  const float* dbhh1 = (const float*)d_in[17];
  const float* fcW   = (const float*)d_in[18];
  const float* fcb   = (const float*)d_in[19];
  float* out = (float*)d_out;

  char* ws = (char*)d_ws;
  size_t off = 0;
  auto alloc = [&](size_t bytes) -> void* {
    off = (off + 255) & ~(size_t)255;
    void* p = ws + off;
    off += bytes;
    return p;
  };

  // bf16 transposed, K-concatenated weights
  unsigned short* WtE0 = (unsigned short*)alloc(768ull  * 2048 * 2);
  unsigned short* WtE1 = (unsigned short*)alloc(1024ull * 2048 * 2);
  unsigned short* WtD0 = (unsigned short*)alloc(512ull  * 2048 * 2);
  unsigned short* WtD1 = (unsigned short*)alloc(1024ull * 2048 * 2);
  float* bias0 = (float*)alloc(2048 * 4);
  float* bias1 = (float*)alloc(2048 * 4);
  float* bias2 = (float*)alloc(2048 * 4);
  float* bias3 = (float*)alloc(2048 * 4);

  // state (zeroed each call); unit0 shared enc0/dec0, unit1 shared enc1/dec1
  const size_t stateStart = (off + 255) & ~(size_t)255;
  unsigned short* h0a = (unsigned short*)alloc(256 * 512 * 2);
  unsigned short* h0b = (unsigned short*)alloc(256 * 512 * 2);
  float*          h0f = (float*)alloc(256 * 512 * 4);
  float*          c0  = (float*)alloc(256 * 512 * 4);
  unsigned short* h1a = (unsigned short*)alloc(256 * 512 * 2);
  unsigned short* h1b = (unsigned short*)alloc(256 * 512 * 2);
  float*          h1f = (float*)alloc(256 * 512 * 4);
  float*          c1  = (float*)alloc(256 * 512 * 4);
  int*            sel = (int*)alloc(256 * 4);
  const size_t stateBytes = off - stateStart;

  // --- init: zero states + argmax feedback -------------------------------
  {
    long long nw = (long long)(stateBytes / 4);
    fill_zero_kernel<<<(unsigned)((nw + 255) / 256), 256, 0, stream>>>(
        (unsigned int*)(ws + stateStart), nw);
  }

  // --- weight conversion (bf16, transposed, concat [Wih^T ; Whh^T]) ------
  auto conv = [&](const float* W, unsigned short* Wt, int K, int rowOff) {
    int total = 2048 * K;
    wt_convert_kernel<<<(total + 255) / 256, 256, 0, stream>>>(W, Wt, K, rowOff, total);
  };
  conv(eWih0, WtE0, 256, 0);
  conv(eWhh0, WtE0, 512, 256);
  conv(eWih1, WtE1, 512, 0);
  conv(eWhh1, WtE1, 512, 512);
  conv(dWhh0, WtD0, 512, 0);
  conv(dWih1, WtD1, 512, 0);
  conv(dWhh1, WtD1, 512, 512);
  bias_combine_kernel<<<8, 256, 0, stream>>>(ebih0, ebhh0, bias0);
  bias_combine_kernel<<<8, 256, 0, stream>>>(ebih1, ebhh1, bias1);
  bias_combine_kernel<<<8, 256, 0, stream>>>(dbih0, dbhh0, bias2);
  bias_combine_kernel<<<8, 256, 0, stream>>>(dbih1, dbhh1, bias3);

  const dim3 cgrid(64), cblk(256);

  // --- encoder: 128 steps, layer0 (emb-gather x) then layer1 -------------
  for (int t = 0; t < 128; ++t) {
    const unsigned short* h0p = (t & 1) ? h0b : h0a;
    unsigned short*       h0n = (t & 1) ? h0a : h0b;
    lstm_cell<0><<<cgrid, cblk, 0, stream>>>(WtE0, bias0, nullptr, emb, x, t,
                                             nullptr, nullptr, h0p, c0, h0n, h0f);
    const unsigned short* h1p = (t & 1) ? h1b : h1a;
    unsigned short*       h1n = (t & 1) ? h1a : h1b;
    lstm_cell<1><<<cgrid, cblk, 0, stream>>>(WtE1, bias1, h0n, nullptr, nullptr, 0,
                                             nullptr, nullptr, h1p, c1, h1n, h1f);
  }

  // --- decoder: continues from encoder final states (same buffers) -------
  for (int s = 0; s < 128; ++s) {
    const int u = 128 + s;  // ping-pong parity continues seamlessly
    const unsigned short* h0p = (u & 1) ? h0b : h0a;
    unsigned short*       h0n = (u & 1) ? h0a : h0b;
    lstm_cell<2><<<cgrid, cblk, 0, stream>>>(WtD0, bias2, nullptr, nullptr, nullptr, 0,
                                             dWih0, sel, h0p, c0, h0n, h0f);
    const unsigned short* h1p = (u & 1) ? h1b : h1a;
    unsigned short*       h1n = (u & 1) ? h1a : h1b;
    lstm_cell<1><<<cgrid, cblk, 0, stream>>>(WtD1, bias3, h0n, nullptr, nullptr, 0,
                                             nullptr, nullptr, h1p, c1, h1n, h1f);
    readout_kernel<<<256, 64, 0, stream>>>(h1f, fcW, fcb, out, sel, s);
  }
}
